// LLMRouter_7773890806139
// MI455X (gfx1250) — matrix operand order repaired and checked by reading, not compiled, not run on hardware
//
#include <hip/hip_runtime.h>
#include <hip/hip_bf16.h>
#include <math.h>

// ---------------------------------------------------------------------------
// Shapes (from reference)
// ---------------------------------------------------------------------------
#define N_L   256
#define N_Q   65536
#define D_IN  384
#define D_CTX 192
#define HID   64
#define MAXA  6
#define LOG_VAR2f (-4.6051701859880914f)   // log(0.01)
#define QITER 2                            // query-groups of 16 per wave per block

typedef __attribute__((ext_vector_type(16))) _Float16 v16h;
typedef __attribute__((ext_vector_type(8)))  float    v8f;
typedef __attribute__((ext_vector_type(4)))  float    f4;
typedef __attribute__((ext_vector_type(4)))  unsigned int u4;

__constant__ float c_logfact[8] = {
    0.0f, 0.0f, 0.69314718056f, 1.79175946923f,
    3.17805383035f, 4.78749174278f, 6.57925121201f, 8.52516136107f };

__device__ __forceinline__ v8f wmma_f16(v16h a, v16h b, v8f c) {
    // D = A(16x32 f16) * B(32x16 f16) + C(16x16 f32)
    return __builtin_amdgcn_wmma_f32_16x16x32_f16(
        false, a, false, b, (short)0, c, false, false);
}

// pack 8 contiguous f32 -> halves A[base..base+7]   (regular loads: LDS/global)
__device__ __forceinline__ void pack8(const float* p, v16h& A, int base, float scale) {
    f4 a = *(const f4*)p;
    f4 b = *(const f4*)(p + 4);
    A[base + 0] = (_Float16)(a.x * scale); A[base + 1] = (_Float16)(a.y * scale);
    A[base + 2] = (_Float16)(a.z * scale); A[base + 3] = (_Float16)(a.w * scale);
    A[base + 4] = (_Float16)(b.x * scale); A[base + 5] = (_Float16)(b.y * scale);
    A[base + 6] = (_Float16)(b.z * scale); A[base + 7] = (_Float16)(b.w * scale);
}

// non-temporal variant for the once-streamed contexts matrix (TH=NT)
__device__ __forceinline__ void pack8nt(const float* p, v16h& A, int base) {
    f4 a = __builtin_nontemporal_load((const f4*)p);
    f4 b = __builtin_nontemporal_load((const f4*)(p + 4));
    A[base + 0] = (_Float16)a.x; A[base + 1] = (_Float16)a.y;
    A[base + 2] = (_Float16)a.z; A[base + 3] = (_Float16)a.w;
    A[base + 4] = (_Float16)b.x; A[base + 5] = (_Float16)b.y;
    A[base + 6] = (_Float16)b.z; A[base + 7] = (_Float16)b.w;
}

// ---------------------------------------------------------------------------
// Kernel 0: zero loss accumulators, convert ctx_w -> f16
// ---------------------------------------------------------------------------
__global__ void prep_kernel(const float* __restrict__ ctxw,
                            _Float16* __restrict__ cw16,
                            float* __restrict__ accs) {
    int i = blockIdx.x * 256 + threadIdx.x;
    if (i < HID * D_CTX) cw16[i] = (_Float16)ctxw[i];
    if (i < 2) accs[i] = 0.0f;
}

// ---------------------------------------------------------------------------
// Kernel 1: VAE (one block of 64 threads per llm row).  Produces normalized
// z (llm_emb) as f16 into ws, and atomically accumulates mse / kld sums.
// ---------------------------------------------------------------------------
__global__ __launch_bounds__(64) void vae_kernel(
    const float* __restrict__ llms,
    const float* __restrict__ fc1w,  const float* __restrict__ fc1b,
    const float* __restrict__ fc21w, const float* __restrict__ fc21b,
    const float* __restrict__ fc22w, const float* __restrict__ fc22b,
    const float* __restrict__ fc3w,  const float* __restrict__ fc3b,
    const float* __restrict__ fc4w,  const float* __restrict__ fc4b,
    const float* __restrict__ eps,
    _Float16* __restrict__ lemb, float* __restrict__ accs) {
    __shared__ float sx[D_IN], sh[HID], sz[HID], sh2[HID], sred[HID];
    const int t = threadIdx.x, r = blockIdx.x;

    for (int i = t; i < D_IN; i += 64) sx[i] = llms[r * D_IN + i];
    __syncthreads();

    float a = fc1b[t];
    for (int k = 0; k < D_IN; ++k) a += fc1w[t * D_IN + k] * sx[k];
    sh[t] = fmaxf(a, 0.0f);
    __syncthreads();

    float mu = fc21b[t], lv = fc22b[t];
    for (int j = 0; j < HID; ++j) {
        float hv = sh[j];
        mu += fc21w[t * HID + j] * hv;
        lv += fc22w[t * HID + j] * hv;
    }
    float stdv = __expf(0.5f * lv) * 0.1f;
    float z = mu + eps[r * HID + t] * stdv;
    sz[t] = z;
    // term of KLD mean (before *-0.5): 1 - log(var2) + lv - (mu^2+e^lv)/var2
    float kld_part = 1.0f - LOG_VAR2f + lv - (mu * mu + __expf(lv)) * 100.0f;
    __syncthreads();

    float b = fc3b[t];
    for (int j = 0; j < HID; ++j) b += fc3w[t * HID + j] * sz[j];
    sh2[t] = fmaxf(b, 0.0f);
    __syncthreads();

    float mse_part = 0.0f;
    for (int p = 0; p < D_IN / HID; ++p) {
        int o = t + HID * p;
        float xh = fc4b[o];
        for (int j = 0; j < HID; ++j) xh += fc4w[o * HID + j] * sh2[j];
        float d = xh - sx[o];
        mse_part += d * d;
    }

    // l2 normalize z -> llm_emb f16
    float ss = 0.0f;
    for (int j = 0; j < HID; ++j) ss += sz[j] * sz[j];
    float inv = 1.0f / fmaxf(sqrtf(ss), 1e-12f);
    lemb[r * HID + t] = (_Float16)(z * inv);

    sred[t] = mse_part; __syncthreads();
    for (int s = 32; s > 0; s >>= 1) { if (t < s) sred[t] += sred[t + s]; __syncthreads(); }
    if (t == 0) atomicAdd(&accs[0], sred[0]);
    __syncthreads();
    sred[t] = kld_part; __syncthreads();
    for (int s = 32; s > 0; s >>= 1) { if (t < s) sred[t] += sred[t + s]; __syncthreads(); }
    if (t == 0) atomicAdd(&accs[1], sred[0]);
}

// ---------------------------------------------------------------------------
// Kernel 2: finalize vae_loss scalar
// ---------------------------------------------------------------------------
__global__ void fin_kernel(const float* __restrict__ accs, float* __restrict__ out) {
    out[0] = accs[0] / (float)(N_L * D_IN) - 0.5f * accs[1] / (float)(N_L * HID);
}

// ---------------------------------------------------------------------------
// Kernel 3: main router.  64 threads (2 waves); each wave owns QITER groups
// of 16 queries.  WMMA f16 for ctx projection (16x64 = 4 tiles x 6 k-chunks)
// and for scores vs all 256 llm embeddings (16 tiles x 2 k-chunks), llm_emb
// staged in LDS; softmax + cumsum + 6-step inverse-CDF sampling per query.
// contexts reads and selected_llm writes use non-temporal cache policy.
// ---------------------------------------------------------------------------
__global__ __launch_bounds__(64) void router_kernel(
    const float* __restrict__ ctx, const int* __restrict__ an_i,
    const float* __restrict__ ru, const _Float16* __restrict__ cw16,
    const _Float16* __restrict__ lemb, const float* __restrict__ ctx_b,
    float* __restrict__ osel, float* __restrict__ ologp) {
    __shared__ __align__(32) _Float16 s_llm[N_L * HID];        // 32 KB
    __shared__ __align__(32) float    s_log[2][16 * N_L];      // 32 KB

    const int tid = threadIdx.x;
    const int lane = tid & 31;
    const int w = tid >> 5;
    const int n = lane & 15;       // column / A-row within tile
    const int g = lane >> 4;       // half-wave group

    // cooperative load of llm_emb f16 into LDS (32 KB, b128 stores)
    {
        const u4* src = (const u4*)lemb;
        u4* dst = (u4*)s_llm;
        for (int i = tid; i < (N_L * HID) / 8; i += 64) dst[i] = src[i];
    }
    __syncthreads();

    float* mylog = s_log[w];

    for (int it = 0; it < QITER; ++it) {
        const int q0 = (blockIdx.x * (2 * QITER) + w * QITER + it) * 16;
        __builtin_prefetch(ctx + (size_t)(q0 + 16) * D_CTX, 0, 0);

        // ---- ctx projection: proj[16][64] = ctx[16][192] @ ctx_w.T ----
        // unroll 2 keeps live VGPRs below the 256 boundary (no vgpr-msb churn)
        v8f acc0 = {}, acc1 = {}, acc2 = {}, acc3 = {};
        #pragma unroll 2
        for (int kc = 0; kc < 6; ++kc) {
            v16h A;
            {   // A layout: lane(m=n, g): K = kc*32 + 16g + h (contiguous 16)
                const float* p = ctx + (size_t)(q0 + n) * D_CTX + kc * 32 + 16 * g;
                pack8nt(p,      A, 0);
                pack8nt(p + 8,  A, 8);
            }
            // B layout: lane(n,g) holds K = kc*32 + 16g + h (contiguous 16 halves)
            v16h B0 = *(const v16h*)(cw16 + (size_t)(16 * 0 + n) * D_CTX + kc * 32 + 16 * g);
            v16h B1 = *(const v16h*)(cw16 + (size_t)(16 * 1 + n) * D_CTX + kc * 32 + 16 * g);
            v16h B2 = *(const v16h*)(cw16 + (size_t)(16 * 2 + n) * D_CTX + kc * 32 + 16 * g);
            v16h B3 = *(const v16h*)(cw16 + (size_t)(16 * 3 + n) * D_CTX + kc * 32 + 16 * g);
            acc0 = wmma_f16(A, B0, acc0);
            acc1 = wmma_f16(A, B1, acc1);
            acc2 = wmma_f16(A, B2, acc2);
            acc3 = wmma_f16(A, B3, acc3);
        }
        // add bias, park proj in LDS as [m][64] (D elem: M = r+8g, N = 16t+n)
        {
            float b0 = ctx_b[n], b1 = ctx_b[16 + n], b2 = ctx_b[32 + n], b3 = ctx_b[48 + n];
            #pragma unroll
            for (int r = 0; r < 8; ++r) {
                int m = r + 8 * g;
                mylog[m * HID +  0 + n] = acc0[r] + b0;
                mylog[m * HID + 16 + n] = acc1[r] + b1;
                mylog[m * HID + 32 + n] = acc2[r] + b2;
                mylog[m * HID + 48 + n] = acc3[r] + b3;
            }
        }
        // ---- l2-norm row n, build f16 A operands for scores GEMM ----
        float ss = 0.0f;
        #pragma unroll
        for (int i = 0; i < HID; i += 4) {
            f4 v = *(const f4*)&mylog[n * HID + i];
            ss += v.x * v.x + v.y * v.y + v.z * v.z + v.w * v.w;
        }
        float invn = 1.0f / fmaxf(sqrtf(ss), 1e-12f);
        v16h A2[2];
        #pragma unroll
        for (int kc = 0; kc < 2; ++kc) {
            const float* p = &mylog[n * HID + kc * 32 + 16 * g];
            pack8(p,      A2[kc], 0, invn);
            pack8(p + 8,  A2[kc], 8, invn);
        }
        // ---- scores GEMM vs all 256 llm embeddings (overwrites mylog) ----
        #pragma unroll 1
        for (int t16 = 0; t16 < 16; ++t16) {
            v8f acc = {};
            #pragma unroll
            for (int kc = 0; kc < 2; ++kc) {
                v16h B = *(const v16h*)&s_llm[(16 * t16 + n) * HID + kc * 32 + 16 * g];
                acc = wmma_f16(A2[kc], B, acc);
            }
            #pragma unroll
            for (int r = 0; r < 8; ++r)
                mylog[(r + 8 * g) * N_L + 16 * t16 + n] = acc[r];
        }

        // ---- per-query softmax / cumsum / sampling (all 32 lanes per query) --
        for (int q = 0; q < 16; ++q) {
            const int qg = q0 + q;
            const float* row = &mylog[q * N_L + lane * 8];
            f4 ra = *(const f4*)row, rb = *(const f4*)(row + 4);
            float s0[8] = { ra.x, ra.y, ra.z, ra.w, rb.x, rb.y, rb.z, rb.w };

            float mx = s0[0];
            #pragma unroll
            for (int j = 1; j < 8; ++j) mx = fmaxf(mx, s0[j]);
            #pragma unroll
            for (int d = 16; d > 0; d >>= 1) mx = fmaxf(mx, __shfl_xor(mx, d));

            float pre[8]; float run = 0.0f;
            #pragma unroll
            for (int j = 0; j < 8; ++j) { run += __expf(s0[j] - mx); pre[j] = run; }
            const float tot = run;

            // inclusive wave scan of lane totals -> exclusive prefix + grand sum
            float x = tot;
            #pragma unroll
            for (int d = 1; d < 32; d <<= 1) {
                float y = __shfl_up(x, (unsigned)d);
                if (lane >= d) x += y;
            }
            const float excl = x - tot;
            const float S = __shfl(x, 31);
            const float logS = __logf(S);
            const float hi = excl + tot;     // lane's max cumsum (unnormalized)

            const int an = an_i[qg];
            float cnt[8] = {0, 0, 0, 0, 0, 0, 0, 0};
            #pragma unroll 1
            for (int i = 0; i < MAXA; ++i) {
                const float u = ru[i * N_Q + qg] * S;   // compare vs unnormalized cumsum
                unsigned msk = (unsigned)__ballot(hi > u);
                int sl = msk ? (int)__builtin_ctz(msk) : 0;
                int jsel = 0;
                if (msk) {
                    #pragma unroll
                    for (int j = 0; j < 8; ++j) jsel += (excl + pre[j] <= u) ? 1 : 0;
                    if (jsel > 7) jsel = 7;
                }
                const float add = (i < an) ? 1.0f : 0.0f;
                if (lane == sl) {
                    #pragma unroll
                    for (int j = 0; j < 8; ++j) if (j == jsel) cnt[j] += add;
                }
            }
            // write selected_llm row: streaming store, don't pollute L2 (TH=NT)
            f4* op = (f4*)&osel[(size_t)qg * N_L + lane * 8];
            f4 o0 = { cnt[0], cnt[1], cnt[2], cnt[3] };
            f4 o1 = { cnt[4], cnt[5], cnt[6], cnt[7] };
            __builtin_nontemporal_store(o0, op);
            __builtin_nontemporal_store(o1, op + 1);

            // log-prob:  logfact(an) - sum logfact(cnt) + sum cnt*log(score)
            float lp = 0.0f;
            #pragma unroll
            for (int j = 0; j < 8; ++j) {
                if (cnt[j] > 0.0f)
                    lp += cnt[j] * (s0[j] - mx - logS) - c_logfact[(int)cnt[j]];
            }
            #pragma unroll
            for (int d = 16; d > 0; d >>= 1) lp += __shfl_xor(lp, d);
            if (lane == 0) ologp[qg] = c_logfact[an] + lp;
        }
    }
}

// ---------------------------------------------------------------------------
// Launch
// ---------------------------------------------------------------------------
extern "C" void kernel_launch(void* const* d_in, const int* in_sizes, int n_in,
                              void* d_out, int out_size, void* d_ws, size_t ws_size,
                              hipStream_t stream) {
    const float* llms  = (const float*)d_in[0];
    const float* ctx   = (const float*)d_in[1];
    const int*   an_i  = (const int*)d_in[2];
    // d_in[3] agent_num_float: gammaln(an+1) handled via table
    const float* fc1w  = (const float*)d_in[4];
    const float* fc1b  = (const float*)d_in[5];
    const float* fc21w = (const float*)d_in[6];
    const float* fc21b = (const float*)d_in[7];
    const float* fc22w = (const float*)d_in[8];
    const float* fc22b = (const float*)d_in[9];
    const float* fc3w  = (const float*)d_in[10];
    const float* fc3b  = (const float*)d_in[11];
    const float* fc4w  = (const float*)d_in[12];
    const float* fc4b  = (const float*)d_in[13];
    const float* ctxw  = (const float*)d_in[14];
    const float* ctxb  = (const float*)d_in[15];
    const float* eps   = (const float*)d_in[16];
    const float* ru    = (const float*)d_in[17];

    float* out   = (float*)d_out;
    float* osel  = out;                                  // [65536*256]
    float* ologp = out + (size_t)N_Q * N_L;              // [65536]
    float* ovae  = ologp + N_Q;                          // [1]

    float*     accs = (float*)d_ws;                      // [0]=mse sum, [1]=kld sum
    _Float16*  cw16 = (_Float16*)((char*)d_ws + 64);                 // 64*192 f16
    _Float16*  lemb = (_Float16*)((char*)d_ws + 64 + HID * D_CTX * 2); // 256*64 f16

    prep_kernel<<<(HID * D_CTX + 255) / 256, 256, 0, stream>>>(ctxw, cw16, accs);
    vae_kernel<<<N_L, HID, 0, stream>>>(llms, fc1w, fc1b, fc21w, fc21b,
                                        fc22w, fc22b, fc3w, fc3b, fc4w, fc4b,
                                        eps, lemb, accs);
    fin_kernel<<<1, 1, 0, stream>>>(accs, ovae);
    // 2 waves * QITER groups * 16 queries = 64 queries per block -> 1024 blocks
    router_kernel<<<N_Q / (2 * QITER * 16), 64, 0, stream>>>(
        ctx, an_i, ru, cw16, lemb, ctxb, osel, ologp);
}
